// DGCF_43379169689882
// MI455X (gfx1250) — matrix operand order repaired and checked by reading, not compiled, not run on hardware
//
#include <hip/hip_runtime.h>
#include <hip/hip_bf16.h>
#include <math.h>

typedef __attribute__((ext_vector_type(16))) _Float16 v16h;
typedef __attribute__((ext_vector_type(8)))  float    v8f;

#define EMB    64
#define NFACT  4
#define DFDIM  16
#define NITER  2
#define NLAYERS 2

// ---------------------------------------------------------------------------
// P0: ego = concat(user_emb, item_emb); out = ego (layer-0 term); A = 1
// ---------------------------------------------------------------------------
__global__ void k_init(const float* __restrict__ ue, const float* __restrict__ ie,
                       float* __restrict__ ego, float* __restrict__ out,
                       float* __restrict__ A, int nU64, int nTot64, int e4)
{
    int i = blockIdx.x * blockDim.x + threadIdx.x;
    if (i < nTot64) {
        float v = (i < nU64) ? ue[i] : ie[i - nU64];
        ego[i] = v;
        out[i] = v;
    }
    if (i < e4) A[i] = 1.0f;
}

// ---------------------------------------------------------------------------
// P1: normA = softmax(A, axis=1); deg[h] += normA   (1 thread / edge)
// ---------------------------------------------------------------------------
__global__ void k_softmax_deg(const float* __restrict__ A, const int* __restrict__ h,
                              float* __restrict__ normA, float* __restrict__ deg, int E)
{
    int e = blockIdx.x * blockDim.x + threadIdx.x;
    if (e >= E) return;
    float4 a = reinterpret_cast<const float4*>(A)[e];
    float m  = fmaxf(fmaxf(a.x, a.y), fmaxf(a.z, a.w));
    float ex = __expf(a.x - m), ey = __expf(a.y - m);
    float ez = __expf(a.z - m), ew = __expf(a.w - m);
    float inv = 1.0f / (ex + ey + ez + ew);
    float4 na = make_float4(ex * inv, ey * inv, ez * inv, ew * inv);
    reinterpret_cast<float4*>(normA)[e] = na;
    int hn = h[e];
    atomicAdd(&deg[hn * 4 + 0], na.x);
    atomicAdd(&deg[hn * 4 + 1], na.y);
    atomicAdd(&deg[hn * 4 + 2], na.z);
    atomicAdd(&deg[hn * 4 + 3], na.w);
}

// ---------------------------------------------------------------------------
// P2: deg -> 1/sqrt(clip(deg, 1e-8)) in place
// ---------------------------------------------------------------------------
__global__ void k_dinv(float* __restrict__ deg, int n4)
{
    int i = blockIdx.x * blockDim.x + threadIdx.x;
    if (i >= n4) return;
    deg[i] = rsqrtf(fmaxf(deg[i], 1e-8f));
}

// ---------------------------------------------------------------------------
// P3: fe[h] += ego[t] * (normA * dinv[h] * dinv[t])   (32 lanes / edge)
// lane l handles elements 2l, 2l+1 -> factor f = l>>3 (factor-major emb layout)
// ---------------------------------------------------------------------------
__global__ void k_scatter(const float* __restrict__ ego, const float* __restrict__ normA,
                          const float* __restrict__ dinv,
                          const int* __restrict__ h, const int* __restrict__ t,
                          float* __restrict__ fe, int E)
{
    int gid  = blockIdx.x * blockDim.x + threadIdx.x;
    int e    = gid >> 5;
    if (e >= E) return;
    int lane = gid & 31;
    int f    = lane >> 3;
    int hn = h[e], tn = t[e];
    float w = normA[e * 4 + f] * dinv[hn * 4 + f] * dinv[tn * 4 + f];
    float2 v = reinterpret_cast<const float2*>(ego + (size_t)tn * EMB)[lane];
    float* dst = fe + (size_t)hn * EMB + 2 * lane;
    atomicAdd(dst + 0, v.x * w);
    atomicAdd(dst + 1, v.y * w);
}

// ---------------------------------------------------------------------------
// P4: A[e,f] += dot( l2norm(fe[h])_f , tanh(l2norm(ego[t])_f) )
// 4 edges x 4 factors = 16 pairs per wave, batched dot via WMMA f16->f32.
//   A-operand (16x32 f16, K=0..15 live): lane p (<16) holds all 16 dims of pair p
//   (K=0..15 across v0..v7); lanes >=16 (K=16..31) zero.
//   B-operand (32x16 f16): lane n (<16) holds all 16 tanh dims of pair n
//   (K=0..15 across v0..v7); lanes >=16 (K=16..31) zero.
//   D diagonal: D[m,m] at {lane m, elem m} for m<8, {lane 16+m, elem m-8} else.
// ---------------------------------------------------------------------------
__global__ void __launch_bounds__(256)
k_attn_wmma(const float* __restrict__ fe, const float* __restrict__ ego,
            const int* __restrict__ h, const int* __restrict__ t,
            float* __restrict__ A, int E)
{
    const int lane = threadIdx.x & 31;
    const int wave = (blockIdx.x * blockDim.x + threadIdx.x) >> 5;
    const int p    = lane & 15;               // pair id 0..15
    int e = wave * 4 + (p >> 2);
    const bool valid = (e < E);
    if (!valid) e = E - 1;                    // clamp loads; keep EXEC full
    const int f   = p & 3;
    const bool hi = lane >= 16;

    // head side: full 16-dim load (only low half-lanes use it)
    const float* hp = fe + (size_t)h[e] * EMB + f * DFDIM;
    float hv[16]; float hh = 0.f;
#pragma unroll
    for (int i = 0; i < 16; i += 4) {
        float4 a4 = *reinterpret_cast<const float4*>(hp + i);
        hv[i] = a4.x; hv[i+1] = a4.y; hv[i+2] = a4.z; hv[i+3] = a4.w;
        hh += a4.x*a4.x + a4.y*a4.y + a4.z*a4.z + a4.w*a4.w;
    }
    const float rh = 1.0f / fmaxf(sqrtf(hh), 1e-12f);

    v16h av = {};
    v16h bv = {};
    const int base = hi ? 8 : 0;
#pragma unroll
    for (int i = 0; i < 16; ++i) av[i] = hi ? (_Float16)0.f : (_Float16)(hv[i] * rh);

    if (!hi) {  // only low half-lanes carry B data (K=0..15)
        const float* tp = ego + (size_t)t[e] * EMB + f * DFDIM;
        float tv[16]; float tt = 0.f;
#pragma unroll
        for (int i = 0; i < 16; i += 4) {
            float4 a4 = *reinterpret_cast<const float4*>(tp + i);
            tv[i] = a4.x; tv[i+1] = a4.y; tv[i+2] = a4.z; tv[i+3] = a4.w;
            tt += a4.x*a4.x + a4.y*a4.y + a4.z*a4.z + a4.w*a4.w;
        }
        const float rt = 1.0f / fmaxf(sqrtf(tt), 1e-12f);
#pragma unroll
        for (int i = 0; i < 16; ++i) bv[i] = (_Float16)tanhf(tv[i] * rt);
    }

    v8f c = {};
    v8f d = __builtin_amdgcn_wmma_f32_16x16x32_f16(
        /*neg_a=*/false, av, /*neg_b=*/false, bv,
        /*c_mod=*/(short)0, c, /*reuse_a=*/false, /*reuse_b=*/false);

    // pull diagonal D[p,p] to every lane of pair p
    float xsel = 0.f;
    const int r7 = p & 7;
#pragma unroll
    for (int r = 0; r < 8; ++r) xsel = (r == r7) ? d[r] : xsel;
    const int srcLane = (p < 8) ? p : (16 + p);
    float dot = __shfl(xsel, srcLane, 32);

    if (valid && lane < 16) A[(size_t)e * 4 + f] += dot;  // unique (e,f) per lane
}

// ---------------------------------------------------------------------------
// P5/P6: output accumulation and final scale (mean over 1 + NLAYERS terms)
// ---------------------------------------------------------------------------
__global__ void k_accum(float* __restrict__ out, const float* __restrict__ x, int n)
{
    int i = blockIdx.x * blockDim.x + threadIdx.x;
    if (i < n) out[i] += x[i];
}

__global__ void k_scale(float* __restrict__ out, int n, float s)
{
    int i = blockIdx.x * blockDim.x + threadIdx.x;
    if (i < n) out[i] *= s;
}

// ---------------------------------------------------------------------------
extern "C" void kernel_launch(void* const* d_in, const int* in_sizes, int n_in,
                              void* d_out, int out_size, void* d_ws, size_t ws_size,
                              hipStream_t stream)
{
    const float* ue = (const float*)d_in[0];
    const float* ie = (const float*)d_in[1];
    const int*   h  = (const int*)d_in[2];
    const int*   t  = (const int*)d_in[3];
    float* out = (float*)d_out;

    const int nU  = in_sizes[0] / EMB;
    const int nI  = in_sizes[1] / EMB;
    const int N   = nU + nI;
    const int E   = in_sizes[2];
    const int N64 = N * EMB;
    const int N4  = N * NFACT;
    const int E4  = E * NFACT;

    // workspace: ego | fe | A | normA | deg  (~111 MB for the given sizes)
    float* ws    = (float*)d_ws;
    float* ego   = ws;
    float* fe    = ego   + (size_t)N64;
    float* Abuf  = fe    + (size_t)N64;
    float* normA = Abuf  + (size_t)E4;
    float* deg   = normA + (size_t)E4;

    const int B = 256;
    int initN = (N64 > E4) ? N64 : E4;
    k_init<<<(initN + B - 1) / B, B, 0, stream>>>(ue, ie, ego, out, Abuf,
                                                  nU * EMB, N64, E4);

    for (int layer = 0; layer < NLAYERS; ++layer) {
        for (int it = 0; it < NITER; ++it) {
            hipMemsetAsync(deg, 0, (size_t)N4 * sizeof(float), stream);
            k_softmax_deg<<<(E + B - 1) / B, B, 0, stream>>>(Abuf, h, normA, deg, E);
            k_dinv<<<(N4 + B - 1) / B, B, 0, stream>>>(deg, N4);

            hipMemsetAsync(fe, 0, (size_t)N64 * sizeof(float), stream);
            {
                long long threads = (long long)E * 32;
                k_scatter<<<(int)((threads + B - 1) / B), B, 0, stream>>>(
                    ego, normA, deg, h, t, fe, E);
            }
            // final attention update never affects the output -> skip it
            if (!(layer == NLAYERS - 1 && it == NITER - 1)) {
                int waves = (E + 3) / 4;
                long long threads = (long long)waves * 32;
                k_attn_wmma<<<(int)((threads + B - 1) / B), B, 0, stream>>>(
                    fe, ego, h, t, Abuf, E);
            }
        }
        // ego <- fe (ping-pong), accumulate layer term into out
        float* tmp = ego; ego = fe; fe = tmp;
        k_accum<<<(N64 + B - 1) / B, B, 0, stream>>>(out, ego, N64);
    }
    k_scale<<<(N64 + B - 1) / B, B, 0, stream>>>(out, N64, 1.0f / (1 + NLAYERS));
}
